// VQWav2VecModel_69664369541327
// MI455X (gfx1250) — matrix operand order repaired and checked by reading, not compiled, not run on hardware
//
#include <hip/hip_runtime.h>
#include <math.h>

typedef __attribute__((ext_vector_type(2)))  float  v2f;
typedef __attribute__((ext_vector_type(8)))  float  v8f;
typedef __attribute__((ext_vector_type(16))) __bf16 v16bf;

#define B_   64
#define CIN  8
#define L0_  8192
#define C1_  32
#define L1_  4096
#define ENC_ 64
#define T_   2048
#define NR_  (B_ * T_)   // 131072 rows of z / quant_z

__device__ __forceinline__ v8f wmma_f32(v2f a, v2f b, v8f c) {
  return __builtin_amdgcn_wmma_f32_16x16x4_f32(false, a, false, b, (short)0, c, false, false);
}
__device__ __forceinline__ v8f wmma_bf16(v16bf a, v16bf b, v8f c) {
  return __builtin_amdgcn_wmma_f32_16x16x32_bf16(false, a, false, b, (short)0, c, false, false);
}

// ---------------- conv1: [64,8,8192] -> relu -> [64,32,4096] ----------------
__global__ void conv1_relu(const float* __restrict__ x, const float* __restrict__ w,
                           const float* __restrict__ bias, float* __restrict__ y) {
  int idx = blockIdx.x * 256 + threadIdx.x;       // b*32*4096 + co*4096 + t
  int t  = idx & (L1_ - 1);
  int co = (idx >> 12) & 31;
  int b  = idx >> 17;
  const float* xb = x + b * CIN * L0_;
  const float* wc = w + co * CIN * 5;
  float acc = bias[co];
  int base = 2 * t - 2;
#pragma unroll
  for (int k = 0; k < 5; ++k) {
    int i = base + k;
    if (i >= 0 && i < L0_) {
#pragma unroll
      for (int ci = 0; ci < CIN; ++ci)
        acc = fmaf(xb[ci * L0_ + i], wc[ci * 5 + k], acc);
    }
  }
  y[idx] = fmaxf(acc, 0.f);
}

// -------- conv2 + transpose: [64,32,4096] -> relu -> z[(b*T+t)*64+eo] --------
__global__ void conv2_relu_t(const float* __restrict__ z1, const float* __restrict__ w,
                             const float* __restrict__ bias, float* __restrict__ z) {
  int idx = blockIdx.x * 256 + threadIdx.x;       // (b*T+t)*64 + eo
  int eo = idx & 63;
  int t  = (idx >> 6) & (T_ - 1);
  int b  = idx >> 17;
  const float* zb = z1 + b * C1_ * L1_;
  const float* wc = w + eo * C1_ * 5;
  float acc = bias[eo];
  int base = 2 * t - 2;
#pragma unroll
  for (int k = 0; k < 5; ++k) {
    int i = base + k;
    if (i >= 0 && i < L1_) {
#pragma unroll
      for (int ci = 0; ci < C1_; ++ci)
        acc = fmaf(zb[ci * L1_ + i], wc[ci * 5 + k], acc);
    }
  }
  z[idx] = fmaxf(acc, 0.f);
}

// ---------------- 0.5*||c_n||^2 for the 128 codebook rows ----------------
__global__ void half_norms(const float* __restrict__ cb, float* __restrict__ hn) {
  int n = threadIdx.x;
  float s = 0.f;
  for (int k = 0; k < 64; ++k) { float c = cb[n * 64 + k]; s += c * c; }
  hn[n] = 0.5f * s;
}

// ---- VQ: f32-WMMA score GEMM (16-row M tile per wave) + cross-lane argmax ----
// argmin dist  ==  argmax (z.c - 0.5||c||^2)
__global__ void vq_argmin_wmma(const float* __restrict__ z, const float* __restrict__ cb,
                               const float* __restrict__ hnorm,
                               float* __restrict__ qz, int* __restrict__ codes) {
  int wid = blockIdx.x * 4 + (threadIdx.x >> 5);
  int l = threadIdx.x & 31, lr = l & 15, hf = l >> 4;
  int m0 = wid * 16;

  // A fragments (f32 16x4 layout): lanes 0-15 K={k,k+1}, lanes 16-31 K={k+2,k+3}, M = lr
  v2f a[16];
  const float* arow = z + (m0 + lr) * 64 + 2 * hf;
#pragma unroll
  for (int kk = 0; kk < 16; ++kk) a[kk] = *(const v2f*)(arow + 4 * kk);

  float best[8]; int bidx[8];
#pragma unroll
  for (int v = 0; v < 8; ++v) { best[v] = -INFINITY; bidx[v] = 0; }

  for (int tn = 0; tn < 8; ++tn) {            // 8 N-tiles cover K=128 codes
    int n = tn * 16 + lr;
    const float* brow = cb + n * 64 + 2 * hf; // B[k][n] = codebook[n][k]
    float hc = hnorm[n];
    v8f acc = {0.f,0.f,0.f,0.f,0.f,0.f,0.f,0.f};
#pragma unroll
    for (int kk = 0; kk < 16; ++kk) {
      v2f bb = *(const v2f*)(brow + 4 * kk);
      acc = wmma_f32(a[kk], bb, acc);
    }
#pragma unroll
    for (int v = 0; v < 8; ++v) {
      float s = acc[v] - hc;                  // row m0+v+8*hf, code n
      if (s > best[v]) { best[v] = s; bidx[v] = n; }
    }
  }
  // reduce over the 16 lanes of each half (wave32 xor shuffles, mask<16 stays in half)
#pragma unroll
  for (int v = 0; v < 8; ++v) {
    for (int off = 1; off < 16; off <<= 1) {
      float ov = __shfl_xor(best[v], off, 32);
      int   oi = __shfl_xor(bidx[v], off, 32);
      if (ov > best[v] || (ov == best[v] && oi < bidx[v])) { best[v] = ov; bidx[v] = oi; }
    }
  }
  // broadcast each row's code, emit codes + quant_z (= codebook[code])
#pragma unroll
  for (int r = 0; r < 16; ++r) {
    int c = __shfl(bidx[r & 7], (r >> 3) << 4, 32);
    if (l == 0) codes[m0 + r] = c;
    qz[(m0 + r) * 64 + l]      = cb[c * 64 + l];
    qz[(m0 + r) * 64 + l + 32] = cb[c * 64 + l + 32];
  }
}

// ------- x-projection GEMM: quant_z[131072,64] @ w_ih^T[64,192] + b_ih -------
// output layout [t][b][192] so the GRU scan reads a contiguous 64x192 slab per step
__global__ void xproj_wmma(const float* __restrict__ qz, const float* __restrict__ wih,
                           const float* __restrict__ bih, float* __restrict__ xp) {
  int wid = blockIdx.x * 4 + (threadIdx.x >> 5);
  int l = threadIdx.x & 31, lr = l & 15, hf = l >> 4;
  int m0 = wid * 16;
  v2f a[16];
  const float* arow = qz + (m0 + lr) * 64 + 2 * hf;
#pragma unroll
  for (int kk = 0; kk < 16; ++kk) a[kk] = *(const v2f*)(arow + 4 * kk);
  for (int j = 0; j < 12; ++j) {
    int n = j * 16 + lr;
    const float* brow = wih + n * 64 + 2 * hf;
    float bias = bih[n];
    v8f acc = {0.f,0.f,0.f,0.f,0.f,0.f,0.f,0.f};
#pragma unroll
    for (int kk = 0; kk < 16; ++kk) {
      v2f bb = *(const v2f*)(brow + 4 * kk);
      acc = wmma_f32(a[kk], bb, acc);
    }
#pragma unroll
    for (int v = 0; v < 8; ++v) {
      int m = m0 + v + 8 * hf;                      // m = b*T + t
      int orow = (m & (T_ - 1)) * 64 + (m >> 11);   // -> t*64 + b
      xp[orow * 192 + n] = acc[v] + bias;
    }
  }
}

// ----------------------- GRU sequential scan, 1 workgroup -----------------------
// 16 waves: group g (batch rows 16g..16g+15) x jj (hidden tile). bf16 WMMA keeps the
// per-step dependent chain at 2 WMMAs per gate tile (K=64 in two 16x16x32 ops).
__global__ void __launch_bounds__(512) gru_scan(const float* __restrict__ xp,
                                                const float* __restrict__ whh,
                                                const float* __restrict__ bhh,
                                                float* __restrict__ ctx) {
  __shared__ __align__(32) __bf16 sw[192 * 64];     // w_hh in bf16, row-major [n][k]
  __shared__ float sh[4][16][68];                   // h per batch-group, padded stride
  int tid = threadIdx.x;
  for (int i = tid; i < 192 * 64; i += 512) sw[i] = (__bf16)whh[i];
  for (int i = tid; i < 4 * 16 * 68; i += 512) (&sh[0][0][0])[i] = 0.f;
  __syncthreads();

  int wv = tid >> 5, l = tid & 31, lr = l & 15, hf = l >> 4;
  int g = wv >> 2, jj = wv & 3;
  int rb = g * 16;
  int ng = jj * 16 + lr;                            // this lane's hidden column
  float bhr = bhh[ng], bhz = bhh[64 + ng], bhn = bhh[128 + ng];

  // B fragments (bf16 32x16): lanes 0-15 -> K base+0..15, lanes 16-31 -> K base+16..31
  const __bf16* bwr = &sw[(jj      * 16 + lr) * 64 + 16 * hf];
  const __bf16* bwz = &sw[((jj + 4) * 16 + lr) * 64 + 16 * hf];
  const __bf16* bwn = &sw[((jj + 8) * 16 + lr) * 64 + 16 * hf];
  v16bf Br0 = *(const v16bf*)(bwr);      v16bf Br1 = *(const v16bf*)(bwr + 32);
  v16bf Bz0 = *(const v16bf*)(bwz);      v16bf Bz1 = *(const v16bf*)(bwz + 32);
  v16bf Bn0 = *(const v16bf*)(bwn);      v16bf Bn1 = *(const v16bf*)(bwn + 32);

  for (int t = 0; t < T_; ++t) {
    // A fragments (bf16 16x32): half0 elems = K{0..15}+32s, half1 = K{16..31}+32s
    v16bf A0, A1;
    const float* hrow = &sh[g][lr][0];
#pragma unroll
    for (int e = 0; e < 8; ++e) {
      A0[e]     = (__bf16)hrow[      16 * hf + e];
      A0[8 + e] = (__bf16)hrow[8 +  16 * hf + e];
      A1[e]     = (__bf16)hrow[32 +  16 * hf + e];
      A1[8 + e] = (__bf16)hrow[40 +  16 * hf + e];
    }
    __syncthreads();   // everyone captured old h before anyone writes new h

    v8f aR = {0.f,0.f,0.f,0.f,0.f,0.f,0.f,0.f};
    v8f aZ = aR, aN = aR;
    aR = wmma_bf16(A0, Br0, aR); aR = wmma_bf16(A1, Br1, aR);
    aZ = wmma_bf16(A0, Bz0, aZ); aZ = wmma_bf16(A1, Bz1, aZ);
    aN = wmma_bf16(A0, Bn0, aN); aN = wmma_bf16(A1, Bn1, aN);

    const float* xb = xp + t * (64 * 192) + rb * 192;
#pragma unroll
    for (int v = 0; v < 8; ++v) {
      int lrow = v + 8 * hf;                       // local batch row in group
      const float* xr_p = xb + lrow * 192;
      float xr = xr_p[ng], xz = xr_p[64 + ng], xn = xr_p[128 + ng];
      float r  = 1.f / (1.f + __expf(-(xr + aR[v] + bhr)));
      float zg = 1.f / (1.f + __expf(-(xz + aZ[v] + bhz)));
      float nn = tanhf(xn + r * (aN[v] + bhn));
      float hold = sh[g][lrow][ng];
      float hnew = (1.f - zg) * nn + zg * hold;
      sh[g][lrow][ng] = hnew;
      ctx[((rb + lrow) * T_ + t) * 64 + ng] = hnew;   // ctx row-major [b*T+t][64]
    }
    __syncthreads();   // h updates visible before next step's A build
  }
}

// ------------- output projection: ctx[131072,64] @ proj_w^T[64,64] + b -------------
__global__ void proj_wmma(const float* __restrict__ ctx, const float* __restrict__ pw,
                          const float* __restrict__ pb, float* __restrict__ pred) {
  int wid = blockIdx.x * 4 + (threadIdx.x >> 5);
  int l = threadIdx.x & 31, lr = l & 15, hf = l >> 4;
  int m0 = wid * 16;
  v2f a[16];
  const float* arow = ctx + (m0 + lr) * 64 + 2 * hf;
#pragma unroll
  for (int kk = 0; kk < 16; ++kk) a[kk] = *(const v2f*)(arow + 4 * kk);
  for (int j = 0; j < 4; ++j) {
    int n = j * 16 + lr;
    const float* brow = pw + n * 64 + 2 * hf;
    float bias = pb[n];
    v8f acc = {0.f,0.f,0.f,0.f,0.f,0.f,0.f,0.f};
#pragma unroll
    for (int kk = 0; kk < 16; ++kk) {
      v2f bb = *(const v2f*)(brow + 4 * kk);
      acc = wmma_f32(a[kk], bb, acc);
    }
#pragma unroll
    for (int v = 0; v < 8; ++v) {
      int m = m0 + v + 8 * hf;
      pred[m * 64 + n] = acc[v] + bias;
    }
  }
}

extern "C" void kernel_launch(void* const* d_in, const int* in_sizes, int n_in,
                              void* d_out, int out_size, void* d_ws, size_t ws_size,
                              hipStream_t stream) {
  const float* x   = (const float*)d_in[0];
  const float* c1w = (const float*)d_in[1];
  const float* c1b = (const float*)d_in[2];
  const float* c2w = (const float*)d_in[3];
  const float* c2b = (const float*)d_in[4];
  const float* cb  = (const float*)d_in[5];
  const float* wih = (const float*)d_in[6];
  const float* whh = (const float*)d_in[7];
  const float* bih = (const float*)d_in[8];
  const float* bhh = (const float*)d_in[9];
  const float* pw  = (const float*)d_in[10];
  const float* pb  = (const float*)d_in[11];

  float* out   = (float*)d_out;
  float* qz    = out;                        // quant_z: 8,388,608 f32
  float* pred  = out + 8388608;              // pred:    8,388,608 f32
  int*   codes = (int*)(out + 16777216);     // codes:   131,072 i32 (bit-cast region)

  char*  ws    = (char*)d_ws;                // 128 MiB plan with aliasing:
  float* z1    = (float*)(ws + 0);           // [0,32M)   conv1 out  (dead after conv2)
  float* z     = (float*)(ws + 33554432);    // [32M,64M) conv2 out  (dead after VQ)
  float* hnorm = (float*)(ws + 67108864);    // 128 f32   (dead after VQ)
  float* xp    = (float*)(ws + 0);           // [0,96M)   x-proj, reuses z1/z/hnorm space
  float* ctx   = (float*)(ws + 100663296);   // [96M,128M) GRU outputs

  conv1_relu    <<<32768, 256, 0, stream>>>(x, c1w, c1b, z1);
  conv2_relu_t  <<<32768, 256, 0, stream>>>(z1, c2w, c2b, z);
  half_norms    <<<1, 128, 0, stream>>>(cb, hnorm);
  vq_argmin_wmma<<<2048, 128, 0, stream>>>(z, cb, hnorm, qz, codes);
  xproj_wmma    <<<2048, 128, 0, stream>>>(qz, wih, bih, xp);
  gru_scan      <<<1, 512, 0, stream>>>(xp, whh, bhh, ctx);
  proj_wmma     <<<2048, 128, 0, stream>>>(ctx, pw, pb, pred);
}
